// WFP4A16RefGatedMLPFusedMoE_50534585205019
// MI455X (gfx1250) — compile-verified, matrix operand order checked by reading, not run on hardware
//
#include <hip/hip_runtime.h>
#include <hip/hip_bf16.h>

// ---------------------------------------------------------------------------
// MoE gated-MLP (Mixtral-style top-2 routing), bf16 WMMA compute, fp32 accum.
// Shapes fixed by the reference: T=1024, H=1024, I=2816, E=8, TOP_K=2.
// ---------------------------------------------------------------------------

#define T_ 1024
#define H_ 1024
#define I_ 2816
#define E_ 8
#define TILE_T 32          // tokens per block (two 16-row WMMA halves)
#define WAVES 4            // wave32 x 4 = 128 threads (one wave per SIMD32)

typedef __attribute__((ext_vector_type(16))) __bf16 v16bf;
typedef __attribute__((ext_vector_type(4)))  __bf16 v4bf;
typedef __attribute__((ext_vector_type(8)))  float  v8f;
typedef __attribute__((ext_vector_type(4)))  float  v4f;

union V16U {
    v4bf         q[4];
    unsigned int u[8];
    v16bf        v;
};
union V8BF {
    v4bf  q[2];
    uint4 u4;
};

// Native fp32 -> bf16 conversion (lowers to packed v_cvt_pk_bf16_f32; RNE).
static __device__ __forceinline__ v4bf cvt4(v4f f) {
    return __builtin_convertvector(f, v4bf);
}

// SiLU with hardware reciprocal (v_rcp_f32): result feeds bf16 storage, so
// ~1 ulp rcp error is far below the bf16 rounding error.
static __device__ __forceinline__ float silu_fast(float a) {
    return a * __builtin_amdgcn_rcpf(1.0f + __expf(-a));
}

// Lane-local 16-element K-slice of an A/B fragment from global fp32:
// 8 floats at p[0..7] and 8 at p[16..23] (caller pre-adds (lane>>4)*8).
static __device__ __forceinline__ v16bf load_frag_g(const float* __restrict__ p) {
    const v4f* q0 = (const v4f*)p;
    const v4f* q1 = (const v4f*)(p + 16);
    V16U r;
    r.q[0] = cvt4(q0[0]);
    r.q[1] = cvt4(q0[1]);
    r.q[2] = cvt4(q1[0]);
    r.q[3] = cvt4(q1[1]);
    return r.v;
}

// Same slice from LDS where data is already packed bf16 (16B + 16B chunks).
static __device__ __forceinline__ v16bf load_frag_lds(const __bf16* p) {
    uint4 a = *(const uint4*)p;          // bf16 x8 : K+0..7
    uint4 b = *(const uint4*)(p + 16);   // bf16 x8 : K+16..23
    V16U r;
    r.u[0] = a.x; r.u[1] = a.y; r.u[2] = a.z; r.u[3] = a.w;
    r.u[4] = b.x; r.u[5] = b.y; r.u[6] = b.z; r.u[7] = b.w;
    return r.v;
}

static __device__ __forceinline__ v8f wmma_bf16(v16bf a, v16bf b, v8f c) {
    return __builtin_amdgcn_wmma_f32_16x16x32_bf16(
        /*neg_a=*/false, a, /*neg_b=*/false, b,
        /*c_mod=*/(short)0, c, /*reuse_a=*/false, /*reuse_b=*/false);
}

// ---------------------------------------------------------------------------
// Kernel 0: zero the output accumulator and the per-expert counters.
// ---------------------------------------------------------------------------
__global__ void moe_zero_kernel(float* __restrict__ out, int n, int* __restrict__ cnt) {
    int i = blockIdx.x * blockDim.x + threadIdx.x;
    if (i < n) out[i] = 0.0f;
    if (i < E_) cnt[i] = 0;
}

// ---------------------------------------------------------------------------
// Kernel 1: router. Softmax over E=8 logits, top-2, renormalize, scatter into
// per-expert token lists. Slot order is nondeterministic but the output is
// not: each token's contribution is slot-independent and the two fp32 atomic
// adds per out element commute bitwise.
// ---------------------------------------------------------------------------
__global__ void moe_router_kernel(const float* __restrict__ logits,
                                  int* __restrict__ cnt,
                                  int* __restrict__ tokId,
                                  float* __restrict__ tokW) {
    int t = blockIdx.x * blockDim.x + threadIdx.x;
    if (t >= T_) return;
    float l[E_];
    float m = -3.4e38f;
#pragma unroll
    for (int e = 0; e < E_; ++e) { l[e] = logits[t * E_ + e]; m = fmaxf(m, l[e]); }
#pragma unroll
    for (int e = 0; e < E_; ++e) l[e] = __expf(l[e] - m);   // softmax numerators
    int i1 = 0; float p1 = l[0];
#pragma unroll
    for (int e = 1; e < E_; ++e) if (l[e] > p1) { p1 = l[e]; i1 = e; }
    int i2 = -1; float p2 = -1.0f;
#pragma unroll
    for (int e = 0; e < E_; ++e) if (e != i1 && l[e] > p2) { p2 = l[e]; i2 = e; }
    float inv = 1.0f / (p1 + p2);                            // IEEE div: once/token
    float g1 = p1 * inv, g2 = p2 * inv;
    int s1 = atomicAdd(&cnt[i1], 1);
    tokId[i1 * T_ + s1] = t; tokW[i1 * T_ + s1] = g1;
    int s2 = atomicAdd(&cnt[i2], 1);
    tokId[i2 * T_ + s2] = t; tokW[i2 * T_ + s2] = g2;
}

// ---------------------------------------------------------------------------
// Kernel 2: fused gated MLP for one tile of 32 tokens of one expert.
//   Prologue: stage the 32 token rows of x in LDS as bf16 (one conversion).
//   Stage 1:  act[32, I] = silu(x@w1^T) * (x@w3^T)  -> bf16 in LDS
//   Stage 2:  y[32, H]   = act @ w2^T ; out[t,:] += gate * y (fp32 atomics)
// All matrix math via v_wmma_f32_16x16x32_bf16; A-fragments always come from
// LDS pre-packed bf16 (ds_load_b128), only streaming weights convert in-flight.
// LDS: 64 KB (xTile) + 176 KB (act) = 240 KB (< 320 KB/workgroup on CDNA5).
// ---------------------------------------------------------------------------
__global__ void __launch_bounds__(WAVES * 32, 1)
moe_ffn_kernel(const float* __restrict__ x,
               const float* __restrict__ w1,
               const float* __restrict__ w3,
               const float* __restrict__ w2,
               const int* __restrict__ cnt,
               const int* __restrict__ tokId,
               const float* __restrict__ tokW,
               float* __restrict__ out) {
    extern __shared__ char smem[];
    __bf16* xTile  = (__bf16*)smem;                          // [TILE_T][H_]
    __bf16* actLds = (__bf16*)(smem + TILE_T * H_ * 2);      // [TILE_T][I_]
    __shared__ int   sTok[TILE_T];
    __shared__ float sGate[TILE_T];

    const int e    = blockIdx.x >> 5;            // 32 tiles max per expert
    const int tile = blockIdx.x & 31;
    const int n    = cnt[e];
    if (tile * TILE_T >= n) return;              // uniform early-exit (pre-barrier)

    const int tid = threadIdx.x;
    if (tid < TILE_T) {
        int idx = tile * TILE_T + tid;
        if (idx < n) { sTok[tid] = tokId[e * T_ + idx]; sGate[tid] = tokW[e * T_ + idx]; }
        else         { sTok[tid] = 0;                   sGate[tid] = 0.0f; }   // padded
    }
    __syncthreads();

    // -------- Prologue: x rows -> bf16 in LDS (each row split over 4 threads)
    {
        const int row  = tid >> 2;                       // 0..31
        const int part = (tid & 3) * (H_ / 4);           // 0,256,512,768
        const float* src = x + (long)sTok[row] * H_ + part;
        __bf16*      dst = xTile + row * H_ + part;
        for (int c = 0; c < H_ / 4; c += 8) {
            V8BF st;
            st.q[0] = cvt4(*(const v4f*)(src + c));
            st.q[1] = cvt4(*(const v4f*)(src + c + 4));
            *(uint4*)(dst + c) = st.u4;
        }
    }
    __syncthreads();

    const int wave = tid >> 5;
    const int lane = tid & 31;
    const int colv = lane & 15;          // fragment row/col owned by this lane
    const int koff = (lane >> 4) << 3;   // K-chunk offset per ISA layout
    const int mb   = (lane >> 4) << 3;   // C/D row base: lanes 16-31 hold M=8..15

    const __bf16* xr0 = xTile + colv * H_;          // A rows 0..15
    const __bf16* xr1 = xTile + (16 + colv) * H_;   // A rows 16..31
    const float*  w1e = w1 + (long)e * I_ * H_;
    const float*  w3e = w3 + (long)e * I_ * H_;
    const float*  w2e = w2 + (long)e * H_ * I_;

    // ---------------- Stage 1: act = silu(x w1^T) * (x w3^T) ----------------
    for (int it = wave; it < I_ / 16; it += WAVES) {
        const int    irow = it * 16 + colv;                // B column = i index
        const float* b1r  = w1e + (long)irow * H_;
        const float* b3r  = w3e + (long)irow * H_;
        v8f acc1a = {}, acc1b = {}, acc3a = {}, acc3b = {};
        for (int h = 0; h < H_; h += 32) {
            v16bf A0 = load_frag_lds(xr0 + h + koff);
            v16bf A1 = load_frag_lds(xr1 + h + koff);
            v16bf B1 = load_frag_g(b1r + h + koff);
            v16bf B3 = load_frag_g(b3r + h + koff);
            acc1a = wmma_bf16(A0, B1, acc1a);
            acc1b = wmma_bf16(A1, B1, acc1b);
            acc3a = wmma_bf16(A0, B3, acc3a);
            acc3b = wmma_bf16(A1, B3, acc3b);
        }
        const int ibase = it * 16 + colv;                  // N index = i
#pragma unroll
        for (int r = 0; r < 8; ++r) {
            float v0 = silu_fast(acc1a[r]) * acc3a[r];
            actLds[(mb + r) * I_ + ibase] = (__bf16)v0;
            float v1 = silu_fast(acc1b[r]) * acc3b[r];
            actLds[(16 + mb + r) * I_ + ibase] = (__bf16)v1;
        }
    }
    __syncthreads();

    // ---------------- Stage 2: out += gate * (act w2^T) ---------------------
    const __bf16* ar0 = actLds + colv * I_;
    const __bf16* ar1 = actLds + (16 + colv) * I_;
    for (int ht = wave; ht < H_ / 16; ht += WAVES) {
        const int    hrow = ht * 16 + colv;                // B column = h index
        const float* b2r  = w2e + (long)hrow * I_;
        v8f acc0 = {}, acc1 = {};
        for (int i = 0; i < I_; i += 32) {
            v16bf A0 = load_frag_lds(ar0 + i + koff);
            v16bf A1 = load_frag_lds(ar1 + i + koff);
            v16bf B  = load_frag_g(b2r + i + koff);
            acc0 = wmma_bf16(A0, B, acc0);
            acc1 = wmma_bf16(A1, B, acc1);
        }
        const int hbase = ht * 16 + colv;                  // N index = h
#pragma unroll
        for (int r = 0; r < 8; ++r) {
            int m0 = mb + r;
            atomicAdd(&out[(long)sTok[m0] * H_ + hbase], sGate[m0] * acc0[r]);
            int m1 = 16 + m0;
            atomicAdd(&out[(long)sTok[m1] * H_ + hbase], sGate[m1] * acc1[r]);
        }
    }
}

// ---------------------------------------------------------------------------
// Host launcher
// ---------------------------------------------------------------------------
extern "C" void kernel_launch(void* const* d_in, const int* in_sizes, int n_in,
                              void* d_out, int out_size, void* d_ws, size_t ws_size,
                              hipStream_t stream) {
    const float* x      = (const float*)d_in[0];   // [T, H]
    const float* logits = (const float*)d_in[1];   // [T, E]
    const float* w1     = (const float*)d_in[2];   // [E, I, H]
    const float* w3     = (const float*)d_in[3];   // [E, I, H]
    const float* w2     = (const float*)d_in[4];   // [E, H, I]
    float*       out    = (float*)d_out;           // [T, H]

    // workspace layout (≈64.3 KB)
    char* ws    = (char*)d_ws;
    int*  cnt   = (int*)ws;                         // [E]
    int*  tokId = (int*)(ws + 256);                 // [E, T]
    float* tokW = (float*)(ws + 256 + (size_t)E_ * T_ * sizeof(int));  // [E, T]

    moe_zero_kernel<<<(T_ * H_ + 255) / 256, 256, 0, stream>>>(out, T_ * H_, cnt);
    moe_router_kernel<<<(T_ + 255) / 256, 256, 0, stream>>>(logits, cnt, tokId, tokW);

    const size_t ldsBytes = (size_t)TILE_T * (H_ + I_) * sizeof(__bf16); // 240 KB
    moe_ffn_kernel<<<E_ * 32, WAVES * 32, ldsBytes, stream>>>(
        x, w1, w3, w2, cnt, tokId, tokW, out);
}
